// AGCRN_11390253269251
// MI455X (gfx1250) — matrix operand order repaired and checked by reading, not compiled, not run on hardware
//
#include <hip/hip_runtime.h>
#include <hip/hip_bf16.h>
#include <math.h>
#include <stdint.h>

typedef __bf16 bf16;
typedef __attribute__((ext_vector_type(16))) __bf16 v16bf;
typedef __attribute__((ext_vector_type(8)))  float  v8f;
typedef __attribute__((ext_vector_type(4)))  unsigned int u32x4;
typedef __attribute__((ext_vector_type(8)))  int i32x8;
typedef __attribute__((ext_vector_type(4)))  int i32x4;

#define NN    1024   // nodes
#define UU    64     // hidden
#define BATCH 32
#define TT    12
#define FIN   2

// ---------------------------------------------------------------------------
// Generic bf16 NN GEMM:  C[M,Ncols] = A[M,Kd] x B[Kd,Ncols]
// Row-major bf16 A and B. Each wave computes a 16x64 output strip (4 WMMAs
// per 32-deep k-step, A fragment reused). Output to f32 and/or bf16.
// Requires: M%16==0, Kd%32==0, Ncols%64==0, waves%8==0.
// ---------------------------------------------------------------------------
__global__ void gemm_bf16(const bf16* __restrict__ A, const bf16* __restrict__ B,
                          int M, int Kd, int Ncols, int strips_n,
                          float* __restrict__ Cf, bf16* __restrict__ Cb) {
    const int wave  = (int)((blockIdx.x * blockDim.x + threadIdx.x) >> 5);
    const int lane  = (int)(threadIdx.x & 31);
    const int strip = wave % strips_n;
    const int rt    = wave / strips_n;
    const int row0  = rt * 16;
    const int col0  = strip * 64;
    const int half  = lane >> 4;
    const int mr    = lane & 15;

    v8f acc[4] = {};
    for (int k0 = 0; k0 < Kd; k0 += 32) {
        // ---- A fragment: 16x32 bf16, row mr (+row0), K split by lane half
        v16bf a;
        const bf16* arow = A + (size_t)(row0 + mr) * Kd + k0 + (half << 3);
        #pragma unroll
        for (int v = 0; v < 8; ++v) {
            const int kb = ((v >> 2) << 4) + ((v & 3) << 1);
            a[2 * v]     = arow[kb];
            a[2 * v + 1] = arow[kb + 1];
        }
        // ---- B fragments: column mr in lanes, K in table order over elements
        const int krow = k0 + (half << 3);
        const bf16* brow = B + (size_t)krow * Ncols + col0 + mr;
        #pragma unroll
        for (int tn = 0; tn < 4; ++tn) {
            v16bf bm;
            #pragma unroll
            for (int j = 0; j < 16; ++j) bm[j] = brow[(size_t)(((j >> 3) << 4) + (j & 7)) * Ncols + tn * 16];
            acc[tn] = __builtin_amdgcn_wmma_f32_16x16x32_bf16(
                false, a, false, bm, (short)0, acc[tn], false, false);
        }
    }
    // ---- hoisted, branch-free store paths
    if (Cf != nullptr) {
        #pragma unroll
        for (int tn = 0; tn < 4; ++tn) {
            const int col = col0 + tn * 16 + mr;
            #pragma unroll
            for (int rr = 0; rr < 8; ++rr) {
                const int row = row0 + rr + (half << 3);
                Cf[(size_t)row * Ncols + col] = acc[tn][rr];
            }
        }
    }
    if (Cb != nullptr) {
        #pragma unroll
        for (int tn = 0; tn < 4; ++tn) {
            const int col = col0 + tn * 16 + mr;
            #pragma unroll
            for (int rr = 0; rr < 8; ++rr) {
                const int row = row0 + rr + (half << 3);
                Cb[(size_t)row * Ncols + col] = (bf16)acc[tn][rr];
            }
        }
    }
}

// ---------------------------------------------------------------------------
// Per-node batched GEMM + bias + activation (+ fused GRU update).
// One block per node n.  Out[b,o] = [feat(b,:) ; Afeat(b,:)] (2I) x W[n] (2I,O)
//   mode 0: sigmoid -> zr_out[b,n,o]  (O = 2U = 128)
//   mode 1: tanh -> hc, then h = r*h + (1-r)*hc in place (O = U = 64)
// blockDim.x = 2*O (O/16 waves).
// W[n] column slice (Kpad x 16, K-tail zero-filled by TDM OOB) is staged into
// LDS per wave via tensor_load_to_lds; LHS is staged once into LDS so the
// inner loop is pure ds_load + WMMA.
// Dynamic LDS: lhs[BATCH*Kpad] then per-wave slices [Kpad*16] each.
// ---------------------------------------------------------------------------
__global__ void pernode_gemm(const bf16* __restrict__ featN,
                             const bf16* __restrict__ AfeatN,
                             const bf16* __restrict__ W,
                             const float* __restrict__ bias,
                             int I, int O, int Kpad, int mode,
                             float* zr_out, const float* zr_in, float* h) {
    extern __shared__ bf16 smem[];
    bf16* lhs = smem;                         // [BATCH][Kpad]
    bf16* wsm = smem + BATCH * Kpad;          // [nwaves][Kpad][16]

    const int n    = (int)blockIdx.x;
    const int tidx = (int)threadIdx.x;
    const int nthr = (int)blockDim.x;
    const int K2   = 2 * I;
    const int wv   = tidx >> 5;
    const int lane = tidx & 31;
    const int col0 = wv * 16;
    const int half = lane >> 4;
    const int mr   = lane & 15;

    // ---- issue per-wave TDM load: W[n][:, col0:col0+16] -> wsm slice
    bf16* wslice = wsm + wv * Kpad * 16;
    {
        const unsigned lds_off = (unsigned)(uintptr_t)wslice;   // low 32 bits = LDS offset
        const unsigned long long ga =
            (unsigned long long)(uintptr_t)(W + (size_t)n * K2 * O + col0);
        // D# group 0: count=1, lds_addr, global_addr[56:0], type=2
        u32x4 g0;
        g0.x = 1u;
        g0.y = lds_off;
        g0.z = (unsigned)(ga & 0xffffffffu);
        g0.w = (unsigned)((ga >> 32) & 0x01ffffffu) | (2u << 30);
        // D# group 1: data_size=2B; tensor_dim0=16, tensor_dim1=K2 (OOB rows -> 0);
        // tile_dim0=16, tile_dim1=Kpad; tensor_dim0_stride=O
        const unsigned td0 = 16u, td1 = (unsigned)K2;
        const unsigned tl0 = 16u, tl1 = (unsigned)Kpad;
        const unsigned long long s0 = (unsigned long long)O;
        i32x8 g1;
        g1[0] = (int)(1u << 16);                                     // data_size=1 (2 bytes)
        g1[1] = (int)((td0 & 0xffffu) << 16);                        // tensor_dim0[15:0]
        g1[2] = (int)(((td0 >> 16) & 0xffffu) | ((td1 & 0xffffu) << 16));
        g1[3] = (int)(((td1 >> 16) & 0xffffu) | ((tl0 & 0xffffu) << 16));
        g1[4] = (int)((tl1 & 0xffffu));                              // tile_dim1, tile_dim2=0
        g1[5] = (int)(s0 & 0xffffffffu);                             // dim0 stride [31:0]
        g1[6] = (int)((s0 >> 32) & 0xffffu);                         // dim0 stride [47:32]
        g1[7] = 0;
        i32x4 gz4  = {0, 0, 0, 0};
        i32x8 gz8  = {0, 0, 0, 0, 0, 0, 0, 0};
        __builtin_amdgcn_tensor_load_to_lds(g0, g1, gz4, gz4, gz8, 0);
    }

    // ---- stage LHS = [feat ; Afeat] (zero K-tail) while TDM runs
    const bf16* fbase = featN  + (size_t)n * (BATCH * I);
    const bf16* abase = AfeatN + (size_t)n * (BATCH * I);
    for (int idx = tidx; idx < BATCH * Kpad; idx += nthr) {
        const int kk = idx % Kpad;
        const int b  = idx / Kpad;
        bf16 v = (bf16)0.0f;
        if (kk < I)        v = fbase[(size_t)b * I + kk];
        else if (kk < K2)  v = abase[(size_t)b * I + (kk - I)];
        lhs[(size_t)b * Kpad + kk] = v;
    }
    __builtin_amdgcn_s_wait_tensorcnt(0);   // our wave's W slice is in LDS
    __syncthreads();                        // lhs visible to all waves

    // ---- K loop: pure LDS -> WMMA
    v8f acc0 = {}, acc1 = {};
    for (int k0 = 0; k0 < Kpad; k0 += 32) {
        v16bf a0, a1, bm;
        const bf16* ar0 = &lhs[(size_t)mr * Kpad + k0 + (half << 3)];
        const bf16* ar1 = &lhs[(size_t)(mr + 16) * Kpad + k0 + (half << 3)];
        #pragma unroll
        for (int v = 0; v < 8; ++v) {
            const int kb = ((v >> 2) << 4) + ((v & 3) << 1);
            a0[2 * v]     = ar0[kb];
            a0[2 * v + 1] = ar0[kb + 1];
            a1[2 * v]     = ar1[kb];
            a1[2 * v + 1] = ar1[kb + 1];
        }
        const bf16* bp = &wslice[(size_t)(k0 + (half << 3)) * 16 + mr];
        #pragma unroll
        for (int j = 0; j < 16; ++j) bm[j] = bp[(((j >> 3) << 4) + (j & 7)) * 16];
        acc0 = __builtin_amdgcn_wmma_f32_16x16x32_bf16(false, a0, false, bm, (short)0, acc0, false, false);
        acc1 = __builtin_amdgcn_wmma_f32_16x16x32_bf16(false, a1, false, bm, (short)0, acc1, false, false);
    }

    // ---- epilogue: bias + activation (+ fused GRU update)
    const int col = col0 + mr;
    const float bs = bias[(size_t)n * O + col];
    if (mode == 0) {
        #pragma unroll
        for (int rr = 0; rr < 8; ++rr) {
            #pragma unroll
            for (int tb = 0; tb < 2; ++tb) {
                const int b = tb * 16 + rr + (half << 3);
                const float v = (tb == 0 ? acc0[rr] : acc1[rr]) + bs;
                zr_out[((size_t)b * NN + n) * (2 * UU) + col] = 1.0f / (1.0f + __expf(-v));
            }
        }
    } else {
        #pragma unroll
        for (int rr = 0; rr < 8; ++rr) {
            #pragma unroll
            for (int tb = 0; tb < 2; ++tb) {
                const int b = tb * 16 + rr + (half << 3);
                const float v = (tb == 0 ? acc0[rr] : acc1[rr]) + bs;
                const float hc = tanhf(v);
                const float rg = zr_in[((size_t)b * NN + n) * (2 * UU) + UU + col];
                const size_t hi = ((size_t)b * NN + n) * UU + col;
                h[hi] = rg * h[hi] + (1.0f - rg) * hc;
            }
        }
    }
}

// ---------------------------------------------------------------------------
// Row-wise relu+softmax of G[N,N] -> bf16 adjacency
// ---------------------------------------------------------------------------
__global__ void relu_softmax(const float* __restrict__ G, bf16* __restrict__ Abf) {
    const int n   = (int)blockIdx.x;
    const int tid = (int)threadIdx.x;       // 256 threads, 4 elems each
    __shared__ float red[256];
    const float* row = G + (size_t)n * NN;
    float vals[4];
    float vmax = 0.0f;
    #pragma unroll
    for (int j = 0; j < 4; ++j) {
        float g = row[tid + 256 * j];
        vals[j] = g > 0.0f ? g : 0.0f;
        vmax = fmaxf(vmax, vals[j]);
    }
    red[tid] = vmax; __syncthreads();
    for (int s = 128; s > 0; s >>= 1) {
        if (tid < s) red[tid] = fmaxf(red[tid], red[tid + s]);
        __syncthreads();
    }
    vmax = red[0]; __syncthreads();
    float sum = 0.0f;
    #pragma unroll
    for (int j = 0; j < 4; ++j) { vals[j] = __expf(vals[j] - vmax); sum += vals[j]; }
    red[tid] = sum; __syncthreads();
    for (int s = 128; s > 0; s >>= 1) {
        if (tid < s) red[tid] += red[tid + s];
        __syncthreads();
    }
    const float inv = 1.0f / red[0];
    #pragma unroll
    for (int j = 0; j < 4; ++j)
        Abf[(size_t)n * NN + tid + 256 * j] = (bf16)(vals[j] * inv);
}

// ---------------------------------------------------------------------------
// Elementwise helpers
// ---------------------------------------------------------------------------
__global__ void conv_f32_bf16(const float* __restrict__ s, bf16* __restrict__ d, size_t nelem) {
    size_t i = (size_t)blockIdx.x * blockDim.x + threadIdx.x;
    if (i < nelem) d[i] = (bf16)s[i];
}

__global__ void transpose_emb(const float* __restrict__ emb, bf16* __restrict__ ET) {
    size_t i = (size_t)blockIdx.x * blockDim.x + threadIdx.x;   // 64*1024
    if (i < (size_t)UU * NN) {
        const int d = (int)(i / NN), n = (int)(i % NN);
        ET[i] = (bf16)emb[(size_t)n * UU + d];
    }
}

// featN[n][b][i] for layer 1 (I=66):  [x_t | (z*)h1]
__global__ void pack_l1(const float* __restrict__ x, int t,
                        const float* __restrict__ h1, const float* __restrict__ zr,
                        int use_z, bf16* __restrict__ featN) {
    const int I = FIN + UU;
    size_t tid = (size_t)blockIdx.x * blockDim.x + threadIdx.x;
    if (tid >= (size_t)NN * BATCH * I) return;
    const int i = (int)(tid % I);
    const int b = (int)((tid / I) % BATCH);
    const int n = (int)(tid / ((size_t)I * BATCH));
    float v;
    if (i < FIN) {
        v = x[(((size_t)b * TT + t) * NN + n) * FIN + i];
    } else {
        v = h1[((size_t)b * NN + n) * UU + (i - FIN)];
        if (use_z) v *= zr[((size_t)b * NN + n) * (2 * UU) + (i - FIN)];
    }
    featN[tid] = (bf16)v;
}

// featN[n][b][i] for layer 2 (I=128): [h1n | (z*)h2]
__global__ void pack_l2(const float* __restrict__ h1, const float* __restrict__ h2,
                        const float* __restrict__ zr, int use_z, bf16* __restrict__ featN) {
    const int I = 2 * UU;
    size_t tid = (size_t)blockIdx.x * blockDim.x + threadIdx.x;
    if (tid >= (size_t)NN * BATCH * I) return;
    const int i = (int)(tid % I);
    const int b = (int)((tid / I) % BATCH);
    const int n = (int)(tid / ((size_t)I * BATCH));
    float v;
    if (i < UU) {
        v = h1[((size_t)b * NN + n) * UU + i];
    } else {
        v = h2[((size_t)b * NN + n) * UU + (i - UU)];
        if (use_z) v *= zr[((size_t)b * NN + n) * (2 * UU) + (i - UU)];
    }
    featN[tid] = (bf16)v;
}

// bias[n,o] = sum_d emb[n,d] * bp[d,o]
__global__ void bias_gemm(const float* __restrict__ emb, const float* __restrict__ bp,
                          float* __restrict__ out, int O) {
    size_t tid = (size_t)blockIdx.x * blockDim.x + threadIdx.x;
    if (tid >= (size_t)NN * O) return;
    const int n = (int)(tid / O), o = (int)(tid % O);
    float s = 0.0f;
    #pragma unroll 8
    for (int d = 0; d < UU; ++d) s += emb[(size_t)n * UU + d] * bp[(size_t)d * O + o];
    out[tid] = s;
}

// out[b,t,n] = h2[b,n,:] . dec_w[t,:] + dec_b[t]
__global__ void decoder_k(const float* __restrict__ h2, const float* __restrict__ w,
                          const float* __restrict__ bvec, float* __restrict__ out) {
    size_t tid = (size_t)blockIdx.x * blockDim.x + threadIdx.x;
    if (tid >= (size_t)BATCH * TT * NN) return;
    const int n = (int)(tid % NN);
    const int t = (int)((tid / NN) % TT);
    const int b = (int)(tid / ((size_t)NN * TT));
    float s = bvec[t];
    #pragma unroll 8
    for (int c = 0; c < UU; ++c)
        s += h2[((size_t)b * NN + n) * UU + c] * w[(size_t)t * UU + c];
    out[tid] = s;
}

// ---------------------------------------------------------------------------
// Host orchestration
// ---------------------------------------------------------------------------
static inline void launch_gemm(const bf16* A, const bf16* B, int M, int Kd, int Ncols,
                               float* Cf, bf16* Cb, hipStream_t s) {
    const int waves  = (M / 16) * (Ncols / 64);
    const int blocks = waves / 8;
    gemm_bf16<<<blocks, 256, 0, s>>>(A, B, M, Kd, Ncols, Ncols / 64, Cf, Cb);
}

static inline void launch_pernode(const bf16* featN, const bf16* AfeatN, const bf16* W,
                                  const float* bias, int I, int O, int Kpad, int mode,
                                  float* zr_out, const float* zr_in, float* h,
                                  hipStream_t s) {
    const int threads = 2 * O;                       // O/16 waves
    const size_t shbytes = ((size_t)BATCH * Kpad + (size_t)(O / 16) * Kpad * 16) * sizeof(bf16);
    pernode_gemm<<<NN, threads, shbytes, s>>>(featN, AfeatN, W, bias, I, O, Kpad, mode,
                                              zr_out, zr_in, h);
}

static inline size_t alignup(size_t v) { return (v + 255) & ~(size_t)255; }

extern "C" void kernel_launch(void* const* d_in, const int* in_sizes, int n_in,
                              void* d_out, int out_size, void* d_ws, size_t ws_size,
                              hipStream_t stream) {
    const float* x      = (const float*)d_in[0];
    const float* emb    = (const float*)d_in[1];
    const float* zr_wp1 = (const float*)d_in[2];
    const float* zr_bp1 = (const float*)d_in[3];
    const float* h_wp1  = (const float*)d_in[4];
    const float* h_bp1  = (const float*)d_in[5];
    const float* zr_wp2 = (const float*)d_in[6];
    const float* zr_bp2 = (const float*)d_in[7];
    const float* h_wp2  = (const float*)d_in[8];
    const float* h_bp2  = (const float*)d_in[9];
    const float* dec_w  = (const float*)d_in[10];
    const float* dec_b  = (const float*)d_in[11];
    float* out = (float*)d_out;

    const int I1 = FIN + UU;          // 66
    const int I2 = 2 * UU;            // 128
    const int Kpad1 = 160;            // ceil(2*66 / 32) * 32
    const int Kpad2 = 256;            // 2*128
    const int C1zr = 2 * I1 * 2 * UU; // 16896
    const int C1h  = 2 * I1 * UU;     // 8448
    const int C2zr = 2 * I2 * 2 * UU; // 32768
    const int C2h  = 2 * I2 * UU;     // 16384

    // ---- workspace carve-up
    char* p = (char*)d_ws;
    auto take = [&](size_t bytes) { char* r = p; p += alignup(bytes); return r; };
    bf16*  Ebf   = (bf16*)take((size_t)NN * UU * 2);
    bf16*  EbfT  = (bf16*)take((size_t)UU * NN * 2);
    float* G     = (float*)take((size_t)NN * NN * 4);
    bf16*  Abf   = (bf16*)take((size_t)NN * NN * 2);
    bf16*  wpbf  = (bf16*)take((size_t)UU * C2zr * 2);           // reused scratch (max)
    bf16*  W1zr  = (bf16*)take((size_t)NN * C1zr * 2);
    bf16*  W1h   = (bf16*)take((size_t)NN * C1h  * 2);
    bf16*  W2zr  = (bf16*)take((size_t)NN * C2zr * 2);
    bf16*  W2h   = (bf16*)take((size_t)NN * C2h  * 2);
    float* b1zr  = (float*)take((size_t)NN * 2 * UU * 4);
    float* b1h   = (float*)take((size_t)NN * UU * 4);
    float* b2zr  = (float*)take((size_t)NN * 2 * UU * 4);
    float* b2h   = (float*)take((size_t)NN * UU * 4);
    float* h1    = (float*)take((size_t)BATCH * NN * UU * 4);
    float* h2    = (float*)take((size_t)BATCH * NN * UU * 4);
    float* zr    = (float*)take((size_t)BATCH * NN * 2 * UU * 4);
    bf16*  featN = (bf16*)take((size_t)NN * BATCH * I2 * 2);
    bf16*  AfeatN= (bf16*)take((size_t)NN * BATCH * I2 * 2);
    (void)ws_size; (void)n_in; (void)in_sizes; (void)out_size;

    auto cdiv = [](size_t a, size_t b) { return (int)((a + b - 1) / b); };

    // ---- 1) adjacency: A = softmax(relu(E E^T))
    conv_f32_bf16<<<cdiv((size_t)NN * UU, 256), 256, 0, stream>>>(emb, Ebf, (size_t)NN * UU);
    transpose_emb<<<cdiv((size_t)UU * NN, 256), 256, 0, stream>>>(emb, EbfT);
    launch_gemm(Ebf, EbfT, NN, UU, NN, G, nullptr, stream);
    relu_softmax<<<NN, 256, 0, stream>>>(G, Abf);

    // ---- 2) expanded per-node weights W = emb @ wp  (time invariant)
    struct { const float* wp; int cols; bf16* W; } wjobs[4] = {
        { zr_wp1, C1zr, W1zr }, { h_wp1, C1h, W1h },
        { zr_wp2, C2zr, W2zr }, { h_wp2, C2h, W2h } };
    for (int j = 0; j < 4; ++j) {
        size_t ne = (size_t)UU * wjobs[j].cols;
        conv_f32_bf16<<<cdiv(ne, 256), 256, 0, stream>>>(wjobs[j].wp, wpbf, ne);
        launch_gemm(Ebf, wpbf, NN, UU, wjobs[j].cols, nullptr, wjobs[j].W, stream);
    }
    // ---- 3) biases b = emb @ bp
    bias_gemm<<<cdiv((size_t)NN * 2 * UU, 256), 256, 0, stream>>>(emb, zr_bp1, b1zr, 2 * UU);
    bias_gemm<<<cdiv((size_t)NN * UU, 256), 256, 0, stream>>>(emb, h_bp1, b1h, UU);
    bias_gemm<<<cdiv((size_t)NN * 2 * UU, 256), 256, 0, stream>>>(emb, zr_bp2, b2zr, 2 * UU);
    bias_gemm<<<cdiv((size_t)NN * UU, 256), 256, 0, stream>>>(emb, h_bp2, b2h, UU);

    // ---- 4) zero initial states
    (void)hipMemsetAsync(h1, 0, (size_t)BATCH * NN * UU * 4, stream);
    (void)hipMemsetAsync(h2, 0, (size_t)BATCH * NN * UU * 4, stream);

    // ---- 5) recurrence
    for (int t = 0; t < TT; ++t) {
        // layer 1 zr gate
        pack_l1<<<cdiv((size_t)NN * BATCH * I1, 256), 256, 0, stream>>>(x, t, h1, nullptr, 0, featN);
        launch_gemm(Abf, featN, NN, NN, BATCH * I1, nullptr, AfeatN, stream);
        launch_pernode(featN, AfeatN, W1zr, b1zr, I1, 2 * UU, Kpad1, 0, zr, nullptr, nullptr, stream);
        // layer 1 candidate + state update (in place on h1)
        pack_l1<<<cdiv((size_t)NN * BATCH * I1, 256), 256, 0, stream>>>(x, t, h1, zr, 1, featN);
        launch_gemm(Abf, featN, NN, NN, BATCH * I1, nullptr, AfeatN, stream);
        launch_pernode(featN, AfeatN, W1h, b1h, I1, UU, Kpad1, 1, nullptr, zr, h1, stream);
        // layer 2 zr gate (input = new h1)
        pack_l2<<<cdiv((size_t)NN * BATCH * I2, 256), 256, 0, stream>>>(h1, h2, nullptr, 0, featN);
        launch_gemm(Abf, featN, NN, NN, BATCH * I2, nullptr, AfeatN, stream);
        launch_pernode(featN, AfeatN, W2zr, b2zr, I2, 2 * UU, Kpad2, 0, zr, nullptr, nullptr, stream);
        // layer 2 candidate + state update (in place on h2)
        pack_l2<<<cdiv((size_t)NN * BATCH * I2, 256), 256, 0, stream>>>(h1, h2, zr, 1, featN);
        launch_gemm(Abf, featN, NN, NN, BATCH * I2, nullptr, AfeatN, stream);
        launch_pernode(featN, AfeatN, W2h, b2h, I2, UU, Kpad2, 1, nullptr, zr, h2, stream);
    }

    // ---- 6) decoder read-out
    decoder_k<<<cdiv((size_t)BATCH * TT * NN, 256), 256, 0, stream>>>(h2, dec_w, dec_b, out);
}
